// MaskedMultiHeadAttention_16913581211937
// MI455X (gfx1250) — matrix operand order repaired and checked by reading, not compiled, not run on hardware
//
#include <hip/hip_runtime.h>
#include <stdint.h>

typedef __bf16 bf16;
typedef __attribute__((ext_vector_type(16))) __bf16 bf16x16;
typedef __attribute__((ext_vector_type(8)))  float  f32x8;
typedef __attribute__((ext_vector_type(4)))  unsigned int u32x4;
typedef __attribute__((ext_vector_type(8)))  int   i32x8;
typedef __attribute__((ext_vector_type(4)))  int   i32x4;

#define BB 4
#define TT 2048
#define HH 16
#define DD 64
#define CC 1024
#define BHTD (BB * HH * TT * DD)   /* 8388608 elements per q/k/v tensor */
#define MROWS (BB * TT)            /* 8192 GEMM rows */

union FragU { uint4 q[2]; bf16x16 f; };

static __device__ __forceinline__ f32x8 wmma_bf16f32(bf16x16 a, bf16x16 b, f32x8 c) {
  // v_wmma_f32_16x16x32_bf16: D = A(16x32) * B(32x16) + C, f32 accumulate
  return __builtin_amdgcn_wmma_f32_16x16x32_bf16(false, a, false, b, (short)0, c,
                                                 false, false);
}

static __device__ __forceinline__ unsigned int pack2(float a, float b) {
  union { bf16 h[2]; unsigned int u; } z;
  z.h[0] = (bf16)a; z.h[1] = (bf16)b;
  return z.u;
}

// wave id as an SGPR: wave-uniform branches become s_cmp/s_cbranch (EXEC
// untouched -> WMMA's EXEC==all-ones requirement is structurally guaranteed).
static __device__ __forceinline__ int wave_id() {
#if __has_builtin(__builtin_amdgcn_readfirstlane)
  return __builtin_amdgcn_readfirstlane((int)(threadIdx.x >> 5));
#else
  return (int)(threadIdx.x >> 5);
#endif
}

// ---- lane-XOR via ds_swizzle (imm pattern; no index VGPR) ------------------
template <int XM>
static __device__ __forceinline__ float xor_lane(float v) {
#if __has_builtin(__builtin_amdgcn_ds_swizzle)
  return __builtin_bit_cast(
      float, __builtin_amdgcn_ds_swizzle(__builtin_bit_cast(int, v),
                                         (XM << 10) | 0x1F));
#else
  return __shfl_xor(v, XM);
#endif
}
static __device__ __forceinline__ float rmax16(float v) {
  v = fmaxf(v, xor_lane<8>(v)); v = fmaxf(v, xor_lane<4>(v));
  v = fmaxf(v, xor_lane<2>(v)); v = fmaxf(v, xor_lane<1>(v));
  return v;
}
static __device__ __forceinline__ float rsum16(float v) {
  v += xor_lane<8>(v); v += xor_lane<4>(v);
  v += xor_lane<2>(v); v += xor_lane<1>(v);
  return v;
}

// ---- Tensor Data Mover: 2D tile global->LDS with LDS row padding -----------
#if __has_builtin(__builtin_amdgcn_tensor_load_to_lds) && \
    __has_builtin(__builtin_amdgcn_s_wait_tensorcnt)
#define USE_TDM 1
// tile0 elems/row (bf16), tile1 rows, stride0 = global row stride (elems).
// padI/padA per D# group1 encoding; tensor dims == tile dims (tile always
// in-bounds, so OOB logic never triggers).
static __device__ __forceinline__ void tdm_load_2d(unsigned lds_off,
                                                   const void* gptr,
                                                   unsigned tile0, unsigned tile1,
                                                   unsigned long long stride0,
                                                   unsigned padI, unsigned padA) {
  unsigned long long ga = (unsigned long long)gptr;
  u32x4 g0;
  g0[0] = 1u;                                          // count=1, user mode
  g0[1] = lds_off;                                     // lds_addr (bytes)
  g0[2] = (unsigned)ga;                                // global_addr[31:0]
  g0[3] = (unsigned)((ga >> 32) & 0x01FFFFFFu) | 0x80000000u;  // ga[56:32]|type=2
  i32x8 g1;
  g1[0] = (int)((1u << 16) | (1u << 20) | (padI << 22) | (padA << 25));
  g1[1] = (int)((tile0 & 0xFFFFu) << 16);              // tensor_dim0[15:0]@48
  g1[2] = (int)(((tile0 >> 16) & 0xFFFFu) | ((tile1 & 0xFFFFu) << 16));
  g1[3] = (int)(((tile1 >> 16) & 0xFFFFu) | ((tile0 & 0xFFFFu) << 16)); // tile_dim0
  g1[4] = (int)(tile1 & 0xFFFFu);                      // tile_dim1, tile_dim2=0
  g1[5] = (int)(unsigned)stride0;                      // dim0_stride[31:0]
  g1[6] = (int)((stride0 >> 32) & 0xFFFFu);            // dim0_stride[47:32]
  g1[7] = 0;
  i32x4 z = {0, 0, 0, 0};
#if __has_include(<hip/amd_detail/amd_gfx1250_TDM.h>)
  i32x8 z8 = {0, 0, 0, 0, 0, 0, 0, 0};
  __builtin_amdgcn_tensor_load_to_lds(g0, g1, z, z, z8, 0);
#else
  __builtin_amdgcn_tensor_load_to_lds(g0, g1, z, z, 0);
#endif
}
static __device__ __forceinline__ unsigned lds_off_of(const void* p) {
  // generic LDS address: high 32 bits = shared aperture, low 32 = LDS offset
  return (unsigned)(unsigned long long)p;
}
#endif

// ---------------------------------------------------------------------------
// Tiled GEMM: C[M,N] = A[M,K](f32) * W[K,N](f32) + bias, via bf16 WMMA.
// 256 threads = 8 waves; WG tile 128x128; wave tile 64x32 (4x2 of 16x16).
// MODE 0: plain f32 row-major store (proj).
// MODE 1: scatter bf16 to q [B,H,T,D] (pre-scaled by 1/sqrt(D)),
//         k [B,H,T,D], v [B,H,D,T].
// ---------------------------------------------------------------------------
template <int MODE>
__global__ __launch_bounds__(256) void gemm_k(const float* __restrict__ A,
                                              const float* __restrict__ W,
                                              const float* __restrict__ bias,
                                              void* __restrict__ Out,
                                              int N, int K) {
  constexpr int AS = 48;  // LDS row stride (bf16): 32 + pad -> 96B (16B mult)
  constexpr int BS = 48;
  __shared__ __align__(16) bf16 sA[128 * AS];  // [m][k]
  __shared__ __align__(16) bf16 sB[128 * BS];  // [n][k]

  const int tid  = threadIdx.x;
  const int wave = wave_id();
  const int lane = tid & 31;
  const int hi   = lane >> 4, lo = lane & 15;
  const int waveM = (wave & 1) * 64;
  const int waveN = (wave >> 1) * 32;
  const int blockM = blockIdx.y * 128;
  const int blockN = blockIdx.x * 128;

  f32x8 acc[4][2] = {};

  for (int k0 = 0; k0 < K; k0 += 32) {
    __syncthreads();
    // Stage A tile 128x32 (f32 -> bf16): 1024 float4 chunks / 256 threads.
#pragma unroll
    for (int i = 0; i < 4; ++i) {
      int f = tid + i * 256;
      int r = f >> 3, c4 = (f & 7) << 2;
      const float* ap = A + (size_t)(blockM + r) * K + k0 + c4;
      float4 v = *(const float4*)ap;
      if (k0 + 32 < K) __builtin_prefetch(ap + 32, 0, 0);
      uint2 pk;
      pk.x = pack2(v.x, v.y);
      pk.y = pack2(v.z, v.w);
      *(uint2*)&sA[r * AS + c4] = pk;
    }
    // Stage B tile 32x128 transposed into [n][k]; pack k-pairs -> b32 stores.
#pragma unroll
    for (int i = 0; i < 2; ++i) {
      int f = tid + i * 256;                 // 512 items: 16 k-pairs x 32 col4
      int kp = f >> 5, c4 = (f & 31) << 2;
      const float* w0 = W + (size_t)(k0 + 2 * kp) * N + blockN + c4;
      float4 va = *(const float4*)(w0);
      float4 vb = *(const float4*)(w0 + N);
      if (k0 + 32 < K) __builtin_prefetch(w0 + (size_t)32 * N, 0, 0);
      *(unsigned*)&sB[(c4 + 0) * BS + 2 * kp] = pack2(va.x, vb.x);
      *(unsigned*)&sB[(c4 + 1) * BS + 2 * kp] = pack2(va.y, vb.y);
      *(unsigned*)&sB[(c4 + 2) * BS + 2 * kp] = pack2(va.z, vb.z);
      *(unsigned*)&sB[(c4 + 3) * BS + 2 * kp] = pack2(va.w, vb.w);
    }
    __syncthreads();

    // A frag (16x32): lane m = lo; contiguous K [hi*16,+16).
    bf16x16 aF[4], bF[2];
#pragma unroll
    for (int i = 0; i < 4; ++i) {
      FragU u;
      const bf16* p = &sA[(waveM + i * 16 + lo) * AS];
      u.q[0] = *(const uint4*)(p + hi * 16);
      u.q[1] = *(const uint4*)(p + hi * 16 + 8);
      aF[i] = u.f;
    }
    // B frag (32x16): lane n = lo; contiguous K [hi*16,+16).
#pragma unroll
    for (int j = 0; j < 2; ++j) {
      FragU u;
      const bf16* p = &sB[(waveN + j * 16 + lo) * BS + hi * 16];
      u.q[0] = *(const uint4*)(p);
      u.q[1] = *(const uint4*)(p + 8);
      bF[j] = u.f;
    }
#pragma unroll
    for (int i = 0; i < 4; ++i)
#pragma unroll
      for (int j = 0; j < 2; ++j)
        acc[i][j] = wmma_bf16f32(aF[i], bF[j], acc[i][j]);
  }

  // Epilogue. C/D layout: VGPR r -> M = r + 8*hi, N = lo.
#pragma unroll
  for (int i = 0; i < 4; ++i)
#pragma unroll
    for (int j = 0; j < 2; ++j) {
      const int col = blockN + waveN + j * 16 + lo;
      const float bv = bias[col];
      if (MODE == 0) {
        float* O = (float*)Out;
#pragma unroll
        for (int r = 0; r < 8; ++r) {
          int row = blockM + waveM + i * 16 + hi * 8 + r;
          O[(size_t)row * N + col] = acc[i][j][r] + bv;
        }
      } else {
        const int part = col >> 10;
        const int h = (col >> 6) & (HH - 1), d = col & (DD - 1);
        const int row0 = blockM + waveM + i * 16 + hi * 8;  // 8 consecutive rows
        const int b = row0 >> 11, t0 = row0 & (TT - 1);
        if (part < 2) {  // q,k: [B,H,T,D]; q pre-scaled by 1/sqrt(D)
          const float sc = (part == 0) ? 0.125f : 1.0f;
          bf16* O = (bf16*)Out + (size_t)part * BHTD;
#pragma unroll
          for (int r = 0; r < 8; ++r)
            O[((size_t)(b * HH + h) * TT + t0 + r) * DD + d] =
                (bf16)((acc[i][j][r] + bv) * sc);
        } else {         // v: [B,H,D,T] -> 8 consecutive t = one b128 store
          bf16* O = (bf16*)Out + (size_t)2 * BHTD;
          union { bf16 h8[8]; uint4 u; } z;
#pragma unroll
          for (int r = 0; r < 8; ++r) z.h8[r] = (bf16)(acc[i][j][r] + bv);
          *(uint4*)&O[((size_t)(b * HH + h) * DD + d) * TT + t0] = z.u;
        }
      }
    }
}

// ---------------------------------------------------------------------------
// Flash attention (causal). One WG per (b, h, 128-query block); 8 waves,
// each wave owns 16 query rows. K staged [key][d] (stride 72), V (already
// [B,H,D,T] in memory) staged [d][key] (stride 136). TDM DMA when available.
// ---------------------------------------------------------------------------
__global__ __launch_bounds__(256) void attn_k(const bf16* __restrict__ Q,
                                              const bf16* __restrict__ K,
                                              const bf16* __restrict__ V,
                                              float* __restrict__ Y) {
  constexpr int KS = 72;   // [key][d]: 64 + 8 pad  (144B rows)
  constexpr int VS = 136;  // [d][key]: 128 + 8 pad (272B rows)
  constexpr int PS = 48;   // P scratch: 32 + pad   (96B rows)
  __shared__ __align__(16) bf16 sK[128 * KS];
  __shared__ __align__(16) bf16 sVT[64 * VS];
  __shared__ __align__(16) bf16 sP[8 * 16 * PS];

  const int tid  = threadIdx.x;
  const int wave = wave_id();            // SGPR: uniform control flow below
  const int lane = tid & 31;
  const int hi   = lane >> 4, lo = lane & 15;
  const int qb = blockIdx.x & 15;
  const int bh = blockIdx.x >> 4;
  const size_t headK = (size_t)bh * TT * DD;  // same element count for K and V
  const int qRow0 = qb * 128 + wave * 16;

  // Q A-fragments (two 16x32 chunks covering D=64), direct from global.
  // Q was pre-scaled by 1/sqrt(D) in the QKV epilogue.
  bf16x16 qF[2];
  {
    const bf16* qr = Q + headK + (size_t)(qRow0 + lo) * DD;
#pragma unroll
    for (int c = 0; c < 2; ++c) {
      FragU u;
      u.q[0] = *(const uint4*)(qr + c * 32 + hi * 16);
      u.q[1] = *(const uint4*)(qr + c * 32 + hi * 16 + 8);
      qF[c] = u.f;
    }
  }

  float mOld[8], lSum[8];
  f32x8 acc[4] = {};
#pragma unroll
  for (int r = 0; r < 8; ++r) { mOld[r] = -__builtin_inff(); lSum[r] = 0.0f; }

  const int waveMax = qRow0 + 15;
  const int nRounds = qb + 1;
  bf16* pBase = &sP[wave * 16 * PS];

  for (int rd = 0; rd < nRounds; ++rd) {
    const int key0 = rd * 128;
    __syncthreads();
    const bf16* kb = K + headK + (size_t)key0 * DD;   // [key][d] rows of 64
    const bf16* vb = V + headK + key0;                // [d][T] rows, col key0
#ifdef USE_TDM
    if (wave == 0) {
      // K tile: 128 rows x 64 elems, row stride 64; pad 16B per 128B row -> 72
      tdm_load_2d(lds_off_of(sK), kb, 64u, 128u, 64ull, 4u, 3u);
      // V tile: 64 rows x 128 elems, row stride 2048; pad 16B per 256B -> 136
      tdm_load_2d(lds_off_of(sVT), vb, 128u, 64u, (unsigned long long)TT, 5u, 3u);
      __builtin_amdgcn_s_wait_tensorcnt(0);
    }
#else
#pragma unroll
    for (int i = 0; i < 4; ++i) {        // K: 1024 b128 chunks / 256 threads
      int c = tid + i * 256;
      int ky = c >> 3, off = (c & 7) << 3;
      *(uint4*)&sK[ky * KS + off] = *(const uint4*)(kb + ky * DD + off);
    }
#pragma unroll
    for (int i = 0; i < 4; ++i) {        // V: contiguous rows, b128 chunks
      int c = tid + i * 256;
      int d = c >> 4, off = (c & 15) << 3;
      *(uint4*)&sVT[d * VS + off] = *(const uint4*)(vb + (size_t)d * TT + off);
    }
#endif
    __syncthreads();

#pragma unroll 1
    for (int sb = 0; sb < 4; ++sb) {
      const int keyStart = key0 + sb * 32;
      if (keyStart > waveMax) break;     // s_cmp/s_cbranch: wave-uniform skip

      // S = Q * K^T for 32 keys (two 16x16 tiles, K-depth 64 = 2 chunks)
      f32x8 s[2];
#pragma unroll
      for (int t = 0; t < 2; ++t) {
        const bf16* kp = &sK[(sb * 32 + t * 16 + lo) * KS];
        FragU u0, u1;
        u0.q[0] = *(const uint4*)(kp + hi * 16);
        u0.q[1] = *(const uint4*)(kp + hi * 16 + 8);
        u1.q[0] = *(const uint4*)(kp + 32 + hi * 16);
        u1.q[1] = *(const uint4*)(kp + 32 + hi * 16 + 8);
        f32x8 z = {};
        z = wmma_bf16f32(qF[0], u0.f, z);
        z = wmma_bf16f32(qF[1], u1.f, z);
        s[t] = z;
      }

      float sc0[8], sc1[8];
#pragma unroll
      for (int r = 0; r < 8; ++r) { sc0[r] = s[0][r]; sc1[r] = s[1][r]; }

      // Causal mask only near the diagonal (wave-uniform guard): blocks with
      // keyStart+31 <= qRow0 are fully visible -> no cmp/cndmask at all.
      if (keyStart + 31 > qRow0) {
#pragma unroll
        for (int r = 0; r < 8; ++r) {
          const int row = qRow0 + hi * 8 + r;
          if (keyStart + lo > row)      sc0[r] = -__builtin_inff();
          if (keyStart + 16 + lo > row) sc1[r] = -__builtin_inff();
        }
      }

      // Online softmax over the 32-key block.
#pragma unroll
      for (int r = 0; r < 8; ++r) {
        const float mx = rmax16(fmaxf(sc0[r], sc1[r]));
        const float mn = fmaxf(mOld[r], mx);
        const float al = __expf(mOld[r] - mn);
        const float p0 = __expf(sc0[r] - mn);
        const float p1 = __expf(sc1[r] - mn);
        pBase[(hi * 8 + r) * PS + lo]      = (bf16)p0;
        pBase[(hi * 8 + r) * PS + 16 + lo] = (bf16)p1;
        lSum[r] = lSum[r] * al + rsum16(p0 + p1);
        mOld[r] = mn;
#pragma unroll
        for (int j = 0; j < 4; ++j) acc[j][r] *= al;
      }

      // Reload P in A-fragment layout (wave-private LDS, in-order per wave).
      FragU pu;
      pu.q[0] = *(const uint4*)(pBase + lo * PS + hi * 16);
      pu.q[1] = *(const uint4*)(pBase + lo * PS + hi * 16 + 8);
      // O += P(16x32) * V(32x64): 4 B-frags along D.
#pragma unroll
      for (int j = 0; j < 4; ++j) {
        FragU vu;
        const bf16* vp = &sVT[(j * 16 + lo) * VS + sb * 32 + hi * 16];
        vu.q[0] = *(const uint4*)(vp);
        vu.q[1] = *(const uint4*)(vp + 8);
        acc[j] = wmma_bf16f32(pu.f, vu.f, acc[j]);
      }
    }
  }

  // Normalize and write y as f32 in [B, T, C] (GEMM-ready).
  const int b = bh >> 4, h = bh & (HH - 1);
#pragma unroll
  for (int r = 0; r < 8; ++r) {
    const float inv = 1.0f / lSum[r];
    const size_t row = (size_t)b * TT + qRow0 + hi * 8 + r;
#pragma unroll
    for (int j = 0; j < 4; ++j)
      Y[row * CC + h * DD + j * 16 + lo] = acc[j][r] * inv;
  }
}

// ---------------------------------------------------------------------------
extern "C" void kernel_launch(void* const* d_in, const int* in_sizes, int n_in,
                              void* d_out, int out_size, void* d_ws, size_t ws_size,
                              hipStream_t stream) {
  (void)in_sizes; (void)n_in; (void)out_size; (void)ws_size;
  const float* x     = (const float*)d_in[0];
  const float* wqkv  = (const float*)d_in[1];
  const float* bqkv  = (const float*)d_in[2];
  const float* wproj = (const float*)d_in[3];
  const float* bproj = (const float*)d_in[4];

  // Workspace: q,k [B,H,T,D] bf16 (q pre-scaled); v [B,H,D,T] bf16; y f32.
  bf16*  q = (bf16*)d_ws;
  bf16*  k = q + BHTD;
  bf16*  v = k + BHTD;
  float* y = (float*)(v + BHTD);

  gemm_k<1><<<dim3(3 * CC / 128, MROWS / 128), 256, 0, stream>>>(
      x, wqkv, bqkv, (void*)q, 3 * CC, CC);
  attn_k<<<dim3(BB * HH * (TT / 128)), 256, 0, stream>>>(q, k, v, y);
  gemm_k<0><<<dim3(CC / 128, MROWS / 128), 256, 0, stream>>>(
      y, wproj, bproj, d_out, CC, CC);
}
